// MultiScaleSTGATBlock_75015898792627
// MI455X (gfx1250) — matrix operand order repaired and checked by reading, not compiled, not run on hardware
//
#include <hip/hip_runtime.h>

typedef __bf16 bf16;
typedef __bf16 v8bf  __attribute__((ext_vector_type(8)));
typedef __bf16 v16bf __attribute__((ext_vector_type(16)));
typedef float  v8f   __attribute__((ext_vector_type(8)));

#define DEV static __device__ __forceinline__

DEV float bf2f(bf16 h) {
  unsigned short s = __builtin_bit_cast(unsigned short, h);
  unsigned u = ((unsigned)s) << 16;
  return __builtin_bit_cast(float, u);
}
DEV bf16 f2bf(float f) {
  unsigned u = __builtin_bit_cast(unsigned, f);
  unsigned r = (u + 0x7fffu + ((u >> 16) & 1u)) >> 16;
  unsigned short s = (unsigned short)r;
  return __builtin_bit_cast(bf16, s);
}
DEV v16bf cat8(v8bf lo, v8bf hi) {
  return __builtin_shufflevector(lo, hi, 0,1,2,3,4,5,6,7,8,9,10,11,12,13,14,15);
}
// 16-bit A fragment (16x32): lane<16 -> row=lane, K={0..15}; lane>=16 -> row=lane-16, K={16..31}
DEV v16bf load_a_frag(const bf16* src, long row, int ldk, int kbase, int lane) {
  const bf16* p = src + row * (long)ldk + kbase + ((lane >> 4) & 1) * 16;
  const v8bf* q = (const v8bf*)p;
  return cat8(q[0], q[1]);   // +0 elems and +8 elems
}
// packed B fragment: 32 lanes x 16 bf16, lane-major contiguous
DEV v16bf load_b_frag(const bf16* frag, int lane) {
  const v8bf* q = (const v8bf*)(frag + lane * 16);
  return cat8(q[0], q[1]);
}
DEV v8f wmma_bf16(v16bf a, v16bf b, v8f c) {
  return __builtin_amdgcn_wmma_f32_16x16x32_bf16(false, a, false, b, (short)0, c, false, false);
}

// ---------------- constants ----------------
// B=32 Cin=64 T=256 V=25 C=128 G=8 ; padded T for conv = 262 (3 zeros each side)

// ---- repack x: hx[(b*T+t)*V+v][64] (GAT A) and xres[(b*V+v)*T+t][64] (res A) ----
__global__ __launch_bounds__(256) void k_repack_x(const float* __restrict__ x,
                                                  bf16* __restrict__ hx,
                                                  bf16* __restrict__ xres) {
  long idx = (long)blockIdx.x * 256 + threadIdx.x;   // over (b,c,t,v) = x flat order
  int v = (int)(idx % 25); long r = idx / 25;
  int t = (int)(r % 256); r /= 256;
  int c = (int)(r % 64);  int b = (int)(r / 64);
  bf16 val = f2bf(x[idx]);
  hx  [(((long)(b * 256 + t) * 25) + v) * 64 + c] = val;
  xres[(((long)(b * 25 + v) * 256) + t) * 64 + c] = val;
}

// ---- pack GAT weights W2[c][h*32+d] into B fragments (2 kc x 8 nt) ----
__global__ __launch_bounds__(256) void k_pack_gat_w(const float* __restrict__ wg, bf16* __restrict__ dst) {
  int tid = blockIdx.x * 256 + threadIdx.x;   // 8192
  int F = tid >> 9, r = tid & 511, l = r >> 4, j = r & 15;
  int kc = F >> 3, nt = F & 7;
  int col = nt * 16 + (l & 15); int h = col >> 5, d = col & 31;
  int c = kc * 32 + ((l >= 16) ? 16 : 0) + j;
  dst[tid] = f2bf(wg[(h * 64 + c) * 32 + d]);
}

// ---- pack conv weights w[co][ci][dt] into frags F=dt*32+kc*8+nt ----
__global__ __launch_bounds__(256) void k_pack_conv_w(const float* __restrict__ w, bf16* __restrict__ dst, int k) {
  int tid = blockIdx.x * 256 + threadIdx.x;   // k*16384
  int F = tid >> 9; int dt = F >> 5; int rem = F & 31;
  int kc = rem >> 3, nt = rem & 7;
  int r = tid & 511, l = r >> 4, j = r & 15;
  int co = nt * 16 + (l & 15);
  int ci = kc * 32 + ((l >= 16) ? 16 : 0) + j;
  dst[tid] = f2bf(w[(co * 128 + ci) * k + dt]);
}

// ---- pack res weights res_w[co][ci] into frags F=kc*8+nt (kc<2) ----
__global__ __launch_bounds__(256) void k_pack_res_w(const float* __restrict__ w, bf16* __restrict__ dst) {
  int tid = blockIdx.x * 256 + threadIdx.x;   // 8192
  int F = tid >> 9; int kc = F >> 3, nt = F & 7;
  int r = tid & 511, l = r >> 4, j = r & 15;
  int co = nt * 16 + (l & 15);
  int ci = kc * 32 + ((l >= 16) ? 16 : 0) + j;
  dst[tid] = f2bf(w[co * 64 + ci]);
}

// ---- GAT projection GEMM: Wh[n][128] = hx[n][64] @ W2, M=204800 ----
__global__ __launch_bounds__(256) void k_gemm_gat(const bf16* __restrict__ hx,
                                                  const bf16* __restrict__ wf,
                                                  bf16* __restrict__ Wh) {
  int wave = threadIdx.x >> 5, lane = threadIdx.x & 31;
  long mtile = (long)blockIdx.x * 8 + wave;
  long row0 = mtile * 16;
  v8f acc[8];
  #pragma unroll
  for (int i = 0; i < 8; i++) {
    #pragma unroll
    for (int r = 0; r < 8; r++) acc[i][r] = 0.f;
  }
  #pragma unroll
  for (int kc = 0; kc < 2; kc++) {
    v16bf a = load_a_frag(hx, row0 + (lane & 15), 64, kc * 32, lane);
    #pragma unroll
    for (int nt = 0; nt < 8; nt++) {
      v16bf b = load_b_frag(wf + (kc * 8 + nt) * 512, lane);
      acc[nt] = wmma_bf16(a, b, acc[nt]);
    }
  }
  int mrow = (lane >= 16) ? 8 : 0, ncol = lane & 15;
  #pragma unroll
  for (int nt = 0; nt < 8; nt++)
    #pragma unroll
    for (int r = 0; r < 8; r++)
      Wh[(row0 + r + mrow) * 128 + nt * 16 + ncol] = f2bf(acc[nt][r]);
}

// ---- GAT attention (softmax over V=25, ELU), writes xg[b][v][t][128] ----
__global__ __launch_bounds__(256) void k_attention(const bf16* __restrict__ Wh,
                                                   const float* __restrict__ adj,
                                                   const float* __restrict__ a_gat,
                                                   bf16* __restrict__ xg) {
  __shared__ float Whs[800];
  __shared__ float e1s[25], e2s[25];
  int bt = blockIdx.x; int b = bt >> 8; int t = bt & 255;
  int tid = threadIdx.x;
  for (int h = 0; h < 4; ++h) {
    for (int i = tid; i < 800; i += 256) {
      int vv = i >> 5, d = i & 31;
      Whs[i] = bf2f(Wh[((long)bt * 25 + vv) * 128 + h * 32 + d]);
    }
    __syncthreads();
    if (tid < 25) {
      float s1 = 0.f, s2 = 0.f;
      for (int d = 0; d < 32; d++) {
        float w = Whs[tid * 32 + d];
        s1 += w * a_gat[h * 64 + d];
        s2 += w * a_gat[h * 64 + 32 + d];
      }
      e1s[tid] = s1; e2s[tid] = s2;
    }
    __syncthreads();
    for (int i = tid; i < 800; i += 256) {
      int n = i >> 5, d = i & 31;
      float e1 = e1s[n];
      float mx = -3.0e38f;
      for (int m = 0; m < 25; m++) {
        float e = e1 + e2s[m];
        e = (e > 0.f) ? e : 0.2f * e;
        e = (adj[n * 25 + m] > 0.f) ? e : -9.0e15f;
        mx = fmaxf(mx, e);
      }
      float den = 0.f, av = 0.f;
      for (int m = 0; m < 25; m++) {
        float e = e1 + e2s[m];
        e = (e > 0.f) ? e : 0.2f * e;
        e = (adj[n * 25 + m] > 0.f) ? e : -9.0e15f;
        float p = __expf(e - mx);
        den += p; av += p * Whs[m * 32 + d];
      }
      float o = av / den;
      o = (o > 0.f) ? o : (__expf(o) - 1.f);   // ELU
      xg[(((long)(b * 25 + n) * 256) + t) * 128 + h * 32 + d] = f2bf(o);
    }
    __syncthreads();
  }
}

// ---- groupnorm stats: per (b, group of 16 ch) over (16,T,V); input [b][v][t][128] ----
__global__ __launch_bounds__(256) void k_gn_stats(const bf16* __restrict__ in, float* __restrict__ stats) {
  __shared__ float rs[256], rq[256];
  int b = blockIdx.x >> 3, g = blockIdx.x & 7;
  float s = 0.f, q = 0.f;
  for (int i = threadIdx.x; i < 102400; i += 256) {
    int ch = i & 15; int p = i >> 4; int tt = p & 255; int vv = p >> 8;
    float val = bf2f(in[(((long)(b * 25 + vv) * 256) + tt) * 128 + g * 16 + ch]);
    s += val; q += val * val;
  }
  rs[threadIdx.x] = s; rq[threadIdx.x] = q;
  __syncthreads();
  for (int off = 128; off; off >>= 1) {
    if ((int)threadIdx.x < off) { rs[threadIdx.x] += rs[threadIdx.x + off]; rq[threadIdx.x] += rq[threadIdx.x + off]; }
    __syncthreads();
  }
  if (threadIdx.x == 0) {
    const float n = 102400.f;
    float mu = rs[0] / n;
    float var = rq[0] / n - mu * mu;
    stats[blockIdx.x * 2 + 0] = mu;
    stats[blockIdx.x * 2 + 1] = rsqrtf(var + 1e-5f);
  }
}

// ---- per-(b,c) scale/offset from stats + gamma/beta ----
__global__ __launch_bounds__(256) void k_make_scale(const float* __restrict__ stats,
                                                    const float* __restrict__ gamma,
                                                    const float* __restrict__ beta,
                                                    float* __restrict__ s, float* __restrict__ o) {
  int i = blockIdx.x * 256 + threadIdx.x;  // 4096
  int b = i >> 7, c = i & 127, g = c >> 4;
  float mu = stats[(b * 8 + g) * 2], rstd = stats[(b * 8 + g) * 2 + 1];
  float sv = gamma[c] * rstd;
  s[i] = sv; o[i] = beta[c] - mu * sv;
}

// ---- build branch input: relu(gn1(xg)) into padded [bv][262][128] bf16 ----
__global__ __launch_bounds__(256) void k_h_build(const bf16* __restrict__ xg,
                                                 const float* __restrict__ s, const float* __restrict__ o,
                                                 bf16* __restrict__ h) {
  long idx = (long)blockIdx.x * 256 + threadIdx.x;  // 32*25*262*128
  int c = (int)(idx & 127); long r = idx >> 7;
  int tp = (int)(r % 262); long bv = r / 262;
  int b = (int)(bv / 25);
  float val = 0.f;
  if (tp >= 3 && tp < 259) {
    float xv = bf2f(xg[(bv * 256 + (tp - 3)) * 128 + c]);
    val = fmaxf(0.f, xv * s[b * 128 + c] + o[b * 128 + c]);
  }
  h[idx] = f2bf(val);
}

// ---- temporal conv as WMMA GEMM: y[bv][t][co] = bias + sum_dt H(t+dt-pad) @ Wdt ----
// Block: 128(t) x 128(co) tile. Wave: 2 m-tiles x 4 n-tiles (B frags reused by both m-tiles).
__global__ __launch_bounds__(256) void k_conv_wmma(const bf16* __restrict__ h,
                                                   const bf16* __restrict__ wf,
                                                   const float* __restrict__ bias,
                                                   bf16* __restrict__ y, int k, int pad) {
  int wave = threadIdx.x >> 5, lane = threadIdx.x & 31;
  int blk = blockIdx.x;            // bv*2 + th
  int th = blk & 1; int bv = blk >> 1;
  int t0 = th * 128;
  int mp = wave >> 1, nh = wave & 1;
  int ncol = lane & 15, mrow = (lane >= 16) ? 8 : 0;
  v8f acc[2][4];
  #pragma unroll
  for (int i = 0; i < 2; i++)
    #pragma unroll
    for (int j = 0; j < 4; j++) {
      float bval = bias[(nh * 4 + j) * 16 + ncol];
      #pragma unroll
      for (int r = 0; r < 8; r++) acc[i][j][r] = bval;
    }
  const bf16* hb = h + (long)bv * 262 * 128;
  int arow0 = t0 + mp * 32 + (lane & 15) + 3 - pad;
  for (int dt = 0; dt < k; ++dt) {
    if (dt + 1 < k)   // prefetch next dt's weight fragment slab (32KB) -> global_prefetch_b8
      __builtin_prefetch(wf + (long)((dt + 1) * 32) * 512 + (int)threadIdx.x * 64, 0, 3);
    for (int kc = 0; kc < 4; ++kc) {
      v16bf a0 = load_a_frag(hb, arow0 + dt, 128, kc * 32, lane);
      v16bf a1 = load_a_frag(hb, arow0 + 16 + dt, 128, kc * 32, lane);
      #pragma unroll
      for (int j = 0; j < 4; j++) {
        v16bf bfr = load_b_frag(wf + ((dt * 4 + kc) * 8 + nh * 4 + j) * 512, lane);
        acc[0][j] = wmma_bf16(a0, bfr, acc[0][j]);
        acc[1][j] = wmma_bf16(a1, bfr, acc[1][j]);
      }
    }
  }
  bf16* yb = y + (long)bv * 256 * 128;
  #pragma unroll
  for (int i = 0; i < 2; i++)
    #pragma unroll
    for (int j = 0; j < 4; j++) {
      int col = (nh * 4 + j) * 16 + ncol;
      #pragma unroll
      for (int r = 0; r < 8; r++)
        yb[(long)(t0 + mp * 32 + i * 16 + mrow + r) * 128 + col] = f2bf(acc[i][j][r]);
    }
}

// ---- fold gn2 offsets + biases into effective bias for fused stage ----
__global__ __launch_bounds__(256) void k_fold_bias(const float* __restrict__ fus_w,
                                                   const float* __restrict__ fus_b,
                                                   const float* __restrict__ res_b,
                                                   const float* __restrict__ o3, const float* __restrict__ o5,
                                                   const float* __restrict__ o7, float* __restrict__ ebias) {
  int i = blockIdx.x * 256 + threadIdx.x;   // 4096
  int b = i >> 7, co = i & 127;
  float acc = fus_b[co] + res_b[co];
  for (int ci = 0; ci < 128; ++ci) {
    acc += fus_w[co * 384 + ci]       * o3[b * 128 + ci];
    acc += fus_w[co * 384 + 128 + ci] * o5[b * 128 + ci];
    acc += fus_w[co * 384 + 256 + ci] * o7[b * 128 + ci];
  }
  ebias[i] = acc;
}

// ---- fold gn2 scale into fusion weights, pack per-b B fragments ----
__global__ __launch_bounds__(256) void k_fold_frags(const float* __restrict__ fus_w,
                                                    const float* __restrict__ s3, const float* __restrict__ s5,
                                                    const float* __restrict__ s7, bf16* __restrict__ fwb) {
  long idx = (long)blockIdx.x * 256 + threadIdx.x;  // 32*96*512
  int r = (int)(idx & 511); long rem = idx >> 9;
  int F = (int)(rem % 96); int b = (int)(rem / 96);
  int l = r >> 4, j = r & 15;
  int br = F >> 5; int sub = F & 31; int kc = sub >> 3, nt = sub & 7;
  int co = nt * 16 + (l & 15);
  int ci = kc * 32 + ((l >= 16) ? 16 : 0) + j;
  const float* sp = (br == 0) ? s3 : ((br == 1) ? s5 : s7);
  fwb[idx] = f2bf(fus_w[co * 384 + br * 128 + ci] * sp[b * 128 + ci]);
}

// ---- fused final GEMM: z = relu( fus(gn2(y3|y5|y7)) + res(x) + biases ), K=448 ----
// Block: 128(rows) x 128(co). Wave: 2 m-tiles x 4 n-tiles.
__global__ __launch_bounds__(256) void k_final_gemm(const bf16* __restrict__ y3, const bf16* __restrict__ y5,
                                                    const bf16* __restrict__ y7, const bf16* __restrict__ xres,
                                                    const bf16* __restrict__ fwb, const bf16* __restrict__ resf,
                                                    const float* __restrict__ ebias, bf16* __restrict__ z) {
  int wave = threadIdx.x >> 5, lane = threadIdx.x & 31;
  int blk = blockIdx.x;            // bv*2 + th
  int th = blk & 1; int bv = blk >> 1; int b = bv / 25;
  int t0 = th * 128;
  int mp = wave >> 1, nh = wave & 1;
  int ncol = lane & 15, mrow = (lane >= 16) ? 8 : 0;
  v8f acc[2][4];
  #pragma unroll
  for (int i = 0; i < 2; i++)
    #pragma unroll
    for (int j = 0; j < 4; j++) {
      float e = ebias[b * 128 + (nh * 4 + j) * 16 + ncol];
      #pragma unroll
      for (int r = 0; r < 8; r++) acc[i][j][r] = e;
    }
  long arow0 = (long)bv * 256 + t0 + mp * 32 + (lane & 15);
  const bf16* fb = fwb + (long)b * 96 * 512;
  const bf16* ys[3] = {y3, y5, y7};
  for (int br = 0; br < 3; br++) {
    const bf16* yb = ys[br];
    for (int kc = 0; kc < 4; kc++) {
      v16bf a0 = load_a_frag(yb, arow0, 128, kc * 32, lane);
      v16bf a1 = load_a_frag(yb, arow0 + 16, 128, kc * 32, lane);
      #pragma unroll
      for (int j = 0; j < 4; j++) {
        v16bf bfr = load_b_frag(fb + ((br * 4 + kc) * 8 + nh * 4 + j) * 512, lane);
        acc[0][j] = wmma_bf16(a0, bfr, acc[0][j]);
        acc[1][j] = wmma_bf16(a1, bfr, acc[1][j]);
      }
    }
  }
  for (int kc = 0; kc < 2; kc++) {
    v16bf a0 = load_a_frag(xres, arow0, 64, kc * 32, lane);
    v16bf a1 = load_a_frag(xres, arow0 + 16, 64, kc * 32, lane);
    #pragma unroll
    for (int j = 0; j < 4; j++) {
      v16bf bfr = load_b_frag(resf + (kc * 8 + nh * 4 + j) * 512, lane);
      acc[0][j] = wmma_bf16(a0, bfr, acc[0][j]);
      acc[1][j] = wmma_bf16(a1, bfr, acc[1][j]);
    }
  }
  bf16* zb = z + (long)bv * 256 * 128;
  #pragma unroll
  for (int i = 0; i < 2; i++)
    #pragma unroll
    for (int j = 0; j < 4; j++) {
      int col = (nh * 4 + j) * 16 + ncol;
      #pragma unroll
      for (int r = 0; r < 8; r++)
        zb[(long)(t0 + mp * 32 + i * 16 + mrow + r) * 128 + col] = f2bf(fmaxf(acc[i][j][r], 0.f));
    }
}

// ---- final groupnorm apply + transpose to (b,c,t,v) f32 output ----
__global__ __launch_bounds__(256) void k_final_out(const bf16* __restrict__ z,
                                                   const float* __restrict__ s, const float* __restrict__ o,
                                                   float* __restrict__ out) {
  long idx = (long)blockIdx.x * 256 + threadIdx.x;  // over (b,v,t,c)
  int c = (int)(idx & 127); long r = idx >> 7;
  int t = (int)(r & 255); r >>= 8;
  int v = (int)(r % 25); int b = (int)(r / 25);
  float val = bf2f(z[idx]) * s[b * 128 + c] + o[b * 128 + c];
  out[(((long)(b * 128 + c) * 256) + t) * 25 + v] = val;
}

extern "C" void kernel_launch(void* const* d_in, const int* in_sizes, int n_in,
                              void* d_out, int out_size, void* d_ws, size_t ws_size,
                              hipStream_t stream) {
  (void)in_sizes; (void)n_in; (void)out_size; (void)ws_size;
  // dict order: x adj w_gat a_gat fus_w fus_b res_w res_b norm_g norm_b
  //             t3g1 t3b1 t3w t3bb t3g2 t3b2  t5...  t7...
  const float* x     = (const float*)d_in[0];
  const float* adj   = (const float*)d_in[1];
  const float* w_gat = (const float*)d_in[2];
  const float* a_gat = (const float*)d_in[3];
  const float* fus_w = (const float*)d_in[4];
  const float* fus_b = (const float*)d_in[5];
  const float* res_w = (const float*)d_in[6];
  const float* res_b = (const float*)d_in[7];
  const float* norm_g = (const float*)d_in[8];
  const float* norm_b = (const float*)d_in[9];
  const float* tg1[3] = {(const float*)d_in[10], (const float*)d_in[16], (const float*)d_in[22]};
  const float* tb1[3] = {(const float*)d_in[11], (const float*)d_in[17], (const float*)d_in[23]};
  const float* tw [3] = {(const float*)d_in[12], (const float*)d_in[18], (const float*)d_in[24]};
  const float* tbb[3] = {(const float*)d_in[13], (const float*)d_in[19], (const float*)d_in[25]};
  const float* tg2[3] = {(const float*)d_in[14], (const float*)d_in[20], (const float*)d_in[26]};
  const float* tb2[3] = {(const float*)d_in[15], (const float*)d_in[21], (const float*)d_in[27]};
  float* out = (float*)d_out;

  char* W = (char*)d_ws;
  size_t off = 0;
  auto alloc = [&](size_t bytes) -> char* {
    char* p = W + off;
    off += (bytes + 511) & ~(size_t)511;
    return p;
  };
  const size_t NB = 26214400ull;            // 204800*128 elems
  bf16* hx   = (bf16*)alloc(204800ull * 64 * 2);
  bf16* xres = (bf16*)alloc(204800ull * 64 * 2);
  bf16* Wh   = (bf16*)alloc(NB * 2);
  bf16* xg   = (bf16*)alloc(NB * 2);
  bf16* hbr[3]; for (int i = 0; i < 3; i++) hbr[i] = (bf16*)alloc(32ull * 25 * 262 * 128 * 2);
  bf16* ybr[3]; for (int i = 0; i < 3; i++) ybr[i] = (bf16*)alloc(NB * 2);
  bf16* z    = (bf16*)alloc(NB * 2);
  bf16* wfG  = (bf16*)alloc(16 * 512 * 2);
  const int kk[3] = {3, 5, 7};
  bf16* wfC[3]; for (int i = 0; i < 3; i++) wfC[i] = (bf16*)alloc((size_t)kk[i] * 32 * 512 * 2);
  bf16* resf = (bf16*)alloc(16 * 512 * 2);
  bf16* fwb  = (bf16*)alloc(32ull * 96 * 512 * 2);
  float* ebias  = (float*)alloc(4096 * 4);
  float* stats1 = (float*)alloc(512 * 4);
  float* stats2[3]; for (int i = 0; i < 3; i++) stats2[i] = (float*)alloc(512 * 4);
  float* stats3 = (float*)alloc(512 * 4);
  float* s1[3]; float* o1[3];
  for (int i = 0; i < 3; i++) { s1[i] = (float*)alloc(4096 * 4); o1[i] = (float*)alloc(4096 * 4); }
  float* s2[3]; float* o2[3];
  for (int i = 0; i < 3; i++) { s2[i] = (float*)alloc(4096 * 4); o2[i] = (float*)alloc(4096 * 4); }
  float* s3 = (float*)alloc(4096 * 4);
  float* o3 = (float*)alloc(4096 * 4);

  // 1) repacks
  k_repack_x<<<51200, 256, 0, stream>>>(x, hx, xres);
  k_pack_gat_w<<<32, 256, 0, stream>>>(w_gat, wfG);
  for (int i = 0; i < 3; i++)
    k_pack_conv_w<<<kk[i] * 64, 256, 0, stream>>>(tw[i], wfC[i], kk[i]);
  k_pack_res_w<<<32, 256, 0, stream>>>(res_w, resf);

  // 2) GAT projection GEMM (WMMA)
  k_gemm_gat<<<1600, 256, 0, stream>>>(hx, wfG, Wh);

  // 3) attention -> xg
  k_attention<<<8192, 256, 0, stream>>>(Wh, adj, a_gat, xg);

  // 4) gn1 stats (shared by all branches) + per-branch affine
  k_gn_stats<<<256, 256, 0, stream>>>(xg, stats1);
  for (int i = 0; i < 3; i++)
    k_make_scale<<<16, 256, 0, stream>>>(stats1, tg1[i], tb1[i], s1[i], o1[i]);

  // 5) branch inputs (padded) ; 6) temporal convs (WMMA) ; 7) gn2 stats
  for (int i = 0; i < 3; i++) {
    k_h_build<<<83800, 256, 0, stream>>>(xg, s1[i], o1[i], hbr[i]);
    k_conv_wmma<<<1600, 256, 0, stream>>>(hbr[i], wfC[i], tbb[i], ybr[i], kk[i], (kk[i] - 1) / 2);
    k_gn_stats<<<256, 256, 0, stream>>>(ybr[i], stats2[i]);
    k_make_scale<<<16, 256, 0, stream>>>(stats2[i], tg2[i], tb2[i], s2[i], o2[i]);
  }

  // 8) fold gn2 into fusion weights/bias ; fused GEMM (fus + res + relu) (WMMA)
  k_fold_bias<<<16, 256, 0, stream>>>(fus_w, fus_b, res_b, o2[0], o2[1], o2[2], ebias);
  k_fold_frags<<<6144, 256, 0, stream>>>(fus_w, s2[0], s2[1], s2[2], fwb);
  k_final_gemm<<<1600, 256, 0, stream>>>(ybr[0], ybr[1], ybr[2], xres, fwb, resf, ebias, z);

  // 9) final groupnorm + transpose to output layout
  k_gn_stats<<<256, 256, 0, stream>>>(z, stats3);
  k_make_scale<<<16, 256, 0, stream>>>(stats3, norm_g, norm_b, s3, o3);
  k_final_out<<<102400, 256, 0, stream>>>(z, s3, o3, out);
}
